// FeatureAttentionGAT_5308579578125
// MI455X (gfx1250) — matrix operand order repaired and checked by reading, not compiled, not run on hardware
//
#include <hip/hip_runtime.h>
#include <cstring>
#include <cstdint>

// ---------------------------------------------------------------------------
// Types / helpers
// ---------------------------------------------------------------------------
typedef __attribute__((ext_vector_type(16))) __bf16 bf16x16;
typedef __attribute__((ext_vector_type(8)))  float  f32x8;
typedef unsigned short u16;

union FragU {
  bf16x16 v;
  u16     u[16];
  uint4   q[2];
};

static __device__ __forceinline__ u16 f2bf(float x) {
  unsigned int u = __float_as_uint(x);
  u += 0x7FFFu + ((u >> 16) & 1);   // round-to-nearest-even
  return (u16)(u >> 16);
}

static __device__ __forceinline__ float wave_sum(float v) {
#pragma unroll
  for (int o = 16; o > 0; o >>= 1) v += __shfl_xor(v, o, 32);
  return v;
}
static __device__ __forceinline__ float wave_max(float v) {
#pragma unroll
  for (int o = 16; o > 0; o >>= 1) v = fmaxf(v, __shfl_xor(v, o, 32));
  return v;
}

static __device__ __forceinline__ int f2ord(float f) {
  int i = __float_as_int(f);
  return i >= 0 ? i : (i ^ 0x7fffffff);
}
static __device__ __forceinline__ float ord2f(int i) {
  return __int_as_float(i >= 0 ? i : (i ^ 0x7fffffff));
}

// A fragment: 16x32 bf16, row-major source with leading dim `ld` (elements).
// Lane l holds row m = row0 + (l&15); ISA K-groups: g = l>>4,
// elems j<8 -> k = k0 + 8g + j ; j>=8 -> k = k0 + 16 + 8g + (j-8).
static __device__ __forceinline__ bf16x16
loadFragA(const u16* base, int ld, int row0, int k0, int lane) {
  int m = row0 + (lane & 15);
  int g = lane >> 4;
  const u16* p = base + (size_t)m * ld + k0 + 8 * g;
  FragU f;
  f.q[0] = *reinterpret_cast<const uint4*>(p);
  f.q[1] = *reinterpret_cast<const uint4*>(p + 16);
  return f.v;
}

// B fragment: 32x16 bf16 where B(k,n) = W[n][k] (row-major W, ld elems).
// Lane l holds column n = n0 + (l&15); g = l>>4,
// elems j<8 -> k = k0 + 8g + j ; j>=8 -> k = k0 + 16 + 8g + (j-8).
static __device__ __forceinline__ bf16x16
loadFragB(const u16* base, int ld, int n0, int k0, int lane) {
  int n = n0 + (lane & 15);
  int g = lane >> 4;
  const u16* p = base + (size_t)n * ld + k0 + 8 * g;
  FragU f;
  f.q[0] = *reinterpret_cast<const uint4*>(p);
  f.q[1] = *reinterpret_cast<const uint4*>(p + 16);
  return f.v;
}

static __device__ __forceinline__ f32x8 wmma_bf16(bf16x16 a, bf16x16 b, f32x8 c) {
  return __builtin_amdgcn_wmma_f32_16x16x32_bf16(false, a, false, b, (short)0, c,
                                                 false, false);
}

// ---------------------------------------------------------------------------
// Encoder constants / LDS layout
// ---------------------------------------------------------------------------
#define SEQ 33    // CLS + 32 tokens
#define MP  48    // padded rows (3 x 16)
#define DM  128
#define NH  4
#define DH  32

// LDS carve (bytes): xF[48*128]f32, scF[48*48]f32, maskAdd[48]f32,
// xH[48*136], qkvH[48*392], vT[4*32*64], pH[48*64], oH[48*136] (bf16)
#define ENC_SMEM_BYTES ((48*128 + 48*48 + 48) * 4 + \
                        (48*136 + 48*392 + 4*32*64 + 48*64 + 48*136) * 2)

__global__ __launch_bounds__(256) void encoder_kernel(
    const float* __restrict__ xin,            // [N,32,128]
    const unsigned char* __restrict__ mask,   // [N,32]  (True = pad)
    const u16* __restrict__ Wqkv,             // [384,128] bf16
    const u16* __restrict__ Wo,               // [128,128] bf16
    const u16* __restrict__ W1,               // [128,128] bf16
    const u16* __restrict__ W2,               // [128,128] bf16
    const float* __restrict__ bqkv, const float* __restrict__ bo,
    const float* __restrict__ b1,   const float* __restrict__ b2,
    const float* __restrict__ ln1s, const float* __restrict__ ln1b,
    const float* __restrict__ ln2s, const float* __restrict__ ln2b,
    const float* __restrict__ clsp,
    u16* __restrict__ featsH, int featsLd, int colOff)
{
  extern __shared__ char smem_raw[];
  float* xF      = (float*)smem_raw;          // 48*128
  float* scF     = xF + 48 * 128;             // 48*48
  float* maskAdd = scF + 48 * 48;             // 48
  u16*   xH      = (u16*)(maskAdd + 48);      // 48*136
  u16*   qkvH    = xH + 48 * 136;             // 48*392
  u16*   vT      = qkvH + 48 * 392;           // 4*32*64 [head][d][token]
  u16*   pH      = vT + 4 * 32 * 64;          // 48*64
  u16*   oH      = pH + 48 * 64;              // 48*136

  const int tid  = threadIdx.x;
  const int lane = tid & 31;
  const int wv   = tid >> 5;
  const int n    = blockIdx.x;

  const float cls = clsp[0];
  const float* xrow = xin + (size_t)n * 32 * DM;

  // ---- stage 0: load x (+CLS, +pad), masks, zero vT ----
  for (int i = tid; i < MP * DM; i += 256) {
    int r = i >> 7, c = i & 127;
    float v = (r == 0) ? cls : ((r <= 32) ? xrow[(r - 1) * DM + c] : 0.f);
    xF[i] = v;
    xH[r * 136 + c] = f2bf(v);
  }
  if (tid < MP) {
    float ma;
    if (tid == 0)       ma = 0.f;
    else if (tid <= 32) ma = mask[(size_t)n * 32 + (tid - 1)] ? -1e9f : 0.f;
    else                ma = -1e9f;
    maskAdd[tid] = ma;
  }
  for (int i = tid; i < NH * DH * 64; i += 256) vT[i] = 0;
  __syncthreads();

  // ---- stage 1: QKV = x @ Wqkv^T + bqkv  -> qkvH (and vT transposed) ----
  for (int t = wv; t < 72; t += 8) {
    int mt = t % 3, nt = t / 3;
    f32x8 c = {};
#pragma unroll
    for (int kk = 0; kk < 4; ++kk) {
      bf16x16 a = loadFragA(xH, 136, mt * 16, kk * 32, lane);
      bf16x16 b = loadFragB(Wqkv, 128, nt * 16, kk * 32, lane);
      c = wmma_bf16(a, b, c);
    }
    int g = lane >> 4, nn = nt * 16 + (lane & 15);
    float bias = bqkv[nn];
#pragma unroll
    for (int i = 0; i < 8; ++i) {
      int m = mt * 16 + i + 8 * g;
      float v = c[i] + bias;
      qkvH[m * 392 + nn] = f2bf(v);
      if (nn >= 256) {  // V head slice -> transposed store vT[head][d][token]
        int hh = (nn - 256) >> 5, dd = (nn - 256) & 31;
        vT[(hh * DH + dd) * 64 + m] = f2bf(v);
      }
    }
  }
  __syncthreads();

  // ---- stage 2: attention (heads sequential; LDS reused) ----
  const float scale = 0.17677669529663687f; // 1/sqrt(32)
  for (int h = 0; h < NH; ++h) {
    // scores = q_h @ k_h^T * scale + mask
    for (int t = wv; t < 9; t += 8) {
      int mt = t % 3, nt = t / 3;
      bf16x16 a = loadFragA(qkvH, 392, mt * 16, h * 32, lane);
      bf16x16 b = loadFragB(qkvH + 128 + h * 32, 392, nt * 16, 0, lane);
      f32x8 c = {};
      c = wmma_bf16(a, b, c);
      int g = lane >> 4, nn = nt * 16 + (lane & 15);
      float madd = maskAdd[nn];
#pragma unroll
      for (int i = 0; i < 8; ++i) {
        int m = mt * 16 + i + 8 * g;
        scF[m * 48 + nn] = c[i] * scale + madd;
      }
    }
    __syncthreads();
    // softmax rows -> pH (bf16, cols >=48 zero, rows >=33 zero)
    for (int r = wv; r < MP; r += 8) {
      if (r >= SEQ) {
        pH[r * 64 + lane] = 0;
        pH[r * 64 + 32 + lane] = 0;
        continue;
      }
      float v0 = scF[r * 48 + lane];
      float v1 = (lane < 16) ? scF[r * 48 + 32 + lane] : -1e30f;
      float mx = wave_max(fmaxf(v0, v1));
      float e0 = __expf(v0 - mx);
      float e1 = (lane < 16) ? __expf(v1 - mx) : 0.f;
      float s  = wave_sum(e0 + e1);
      float inv = 1.f / s;
      pH[r * 64 + lane]      = f2bf(e0 * inv);
      pH[r * 64 + 32 + lane] = (lane < 16) ? f2bf(e1 * inv) : (u16)0;
    }
    __syncthreads();
    // O_h = P @ V_h  (B from transposed vT -> contiguous)
    for (int t = wv; t < 6; t += 8) {
      int mt = t % 3, nt = t / 3;
      f32x8 c = {};
#pragma unroll
      for (int kk = 0; kk < 2; ++kk) {
        bf16x16 a = loadFragA(pH, 64, mt * 16, kk * 32, lane);
        bf16x16 b = loadFragB(vT + h * DH * 64, 64, nt * 16, kk * 32, lane);
        c = wmma_bf16(a, b, c);
      }
      int g = lane >> 4, nn = nt * 16 + (lane & 15);
#pragma unroll
      for (int i = 0; i < 8; ++i) {
        int m = mt * 16 + i + 8 * g;
        oH[m * 136 + h * 32 + nn] = f2bf(c[i]);
      }
    }
    __syncthreads();
  }

  // ---- stage 3: x = LN1(x + o @ Wo^T + bo) ----
  for (int t = wv; t < 24; t += 8) {
    int mt = t % 3, nt = t / 3;
    f32x8 c = {};
#pragma unroll
    for (int kk = 0; kk < 4; ++kk) {
      bf16x16 a = loadFragA(oH, 136, mt * 16, kk * 32, lane);
      bf16x16 b = loadFragB(Wo, 128, nt * 16, kk * 32, lane);
      c = wmma_bf16(a, b, c);
    }
    int g = lane >> 4, nn = nt * 16 + (lane & 15);
    float bias = bo[nn];
#pragma unroll
    for (int i = 0; i < 8; ++i) {
      int m = mt * 16 + i + 8 * g;
      xF[m * 128 + nn] += c[i] + bias;
    }
  }
  __syncthreads();
  for (int r = wv; r < MP; r += 8) {
    float vals[4]; float acc = 0.f;
#pragma unroll
    for (int j = 0; j < 4; ++j) { vals[j] = xF[r * 128 + lane + 32 * j]; acc += vals[j]; }
    float mean = wave_sum(acc) * (1.f / 128.f);
    float va = 0.f;
#pragma unroll
    for (int j = 0; j < 4; ++j) { float d = vals[j] - mean; va += d * d; }
    float inv = rsqrtf(wave_sum(va) * (1.f / 128.f) + 1e-5f);
#pragma unroll
    for (int j = 0; j < 4; ++j) {
      int cc = lane + 32 * j;
      float o = (vals[j] - mean) * inv * ln1s[cc] + ln1b[cc];
      xF[r * 128 + cc] = o;
      xH[r * 136 + cc] = f2bf(o);
    }
  }
  __syncthreads();

  // ---- stage 4: FFN: h = relu(x@W1^T+b1); x = LN2(x + h@W2^T+b2) ----
  for (int t = wv; t < 24; t += 8) {
    int mt = t % 3, nt = t / 3;
    f32x8 c = {};
#pragma unroll
    for (int kk = 0; kk < 4; ++kk) {
      bf16x16 a = loadFragA(xH, 136, mt * 16, kk * 32, lane);
      bf16x16 b = loadFragB(W1, 128, nt * 16, kk * 32, lane);
      c = wmma_bf16(a, b, c);
    }
    int g = lane >> 4, nn = nt * 16 + (lane & 15);
    float bias = b1[nn];
#pragma unroll
    for (int i = 0; i < 8; ++i) {
      int m = mt * 16 + i + 8 * g;
      oH[m * 136 + nn] = f2bf(fmaxf(c[i] + bias, 0.f));
    }
  }
  __syncthreads();
  for (int t = wv; t < 24; t += 8) {
    int mt = t % 3, nt = t / 3;
    f32x8 c = {};
#pragma unroll
    for (int kk = 0; kk < 4; ++kk) {
      bf16x16 a = loadFragA(oH, 136, mt * 16, kk * 32, lane);
      bf16x16 b = loadFragB(W2, 128, nt * 16, kk * 32, lane);
      c = wmma_bf16(a, b, c);
    }
    int g = lane >> 4, nn = nt * 16 + (lane & 15);
    float bias = b2[nn];
#pragma unroll
    for (int i = 0; i < 8; ++i) {
      int m = mt * 16 + i + 8 * g;
      xF[m * 128 + nn] += c[i] + bias;
    }
  }
  __syncthreads();
  for (int r = wv; r < MP; r += 8) {
    float vals[4]; float acc = 0.f;
#pragma unroll
    for (int j = 0; j < 4; ++j) { vals[j] = xF[r * 128 + lane + 32 * j]; acc += vals[j]; }
    float mean = wave_sum(acc) * (1.f / 128.f);
    float va = 0.f;
#pragma unroll
    for (int j = 0; j < 4; ++j) { float d = vals[j] - mean; va += d * d; }
    float inv = rsqrtf(wave_sum(va) * (1.f / 128.f) + 1e-5f);
#pragma unroll
    for (int j = 0; j < 4; ++j) {
      int cc = lane + 32 * j;
      xF[r * 128 + cc] = (vals[j] - mean) * inv * ln2s[cc] + ln2b[cc];
    }
  }
  __syncthreads();

  // ---- stage 5: CLS pooling -> feats (bf16) ----
  if (tid < DM)
    featsH[(size_t)n * featsLd + colOff + tid] = f2bf(xF[tid]);
}

// ---------------------------------------------------------------------------
// Generic tiled WMMA GEMM: C[M][Nc] = A[M][K] @ W[Nc][K]^T  (bf16 in, f32 out)
// One 16x16 tile per wave.
// ---------------------------------------------------------------------------
__global__ __launch_bounds__(256) void gemm_bf16_kernel(
    const u16* __restrict__ A, const u16* __restrict__ W,
    float* __restrict__ C, int M, int Nc, int K)
{
  int lane = threadIdx.x & 31, wv = threadIdx.x >> 5;
  int tilesN = Nc >> 4;
  int tile = blockIdx.x * 8 + wv;
  int total = (M >> 4) * tilesN;
  if (tile >= total) return;
  int mt = tile / tilesN, nt = tile % tilesN;
  f32x8 c = {};
  for (int k0 = 0; k0 < K; k0 += 32) {
    bf16x16 a = loadFragA(A, K, mt * 16, k0, lane);
    bf16x16 b = loadFragB(W, K, nt * 16, k0, lane);
    c = wmma_bf16(a, b, c);
  }
  int g = lane >> 4, nn = nt * 16 + (lane & 15);
#pragma unroll
  for (int i = 0; i < 8; ++i) {
    int m = mt * 16 + i + 8 * g;
    C[(size_t)m * Nc + nn] = c[i];
  }
}

// ---------------------------------------------------------------------------
// GAT support kernels
// ---------------------------------------------------------------------------
__global__ void alpha_kernel(const float* __restrict__ Hm,
                             const float* __restrict__ aS, const float* __restrict__ aD,
                             float* __restrict__ alphaS, float* __restrict__ alphaD,
                             int N, int C)
{
  int lane = threadIdx.x & 31;
  int node = (int)((blockIdx.x * blockDim.x + threadIdx.x) >> 5);
  if (node >= N) return;
  const float* hp = Hm + (size_t)node * C;
  float s = 0.f, d = 0.f;
  for (int j = lane; j < C; j += 32) { float v = hp[j]; s += v * aS[j]; d += v * aD[j]; }
  s = wave_sum(s); d = wave_sum(d);
  if (lane == 0) { alphaS[node] = s; alphaD[node] = d; }
}

__global__ void fill_f32(float* p, float v, int n) {
  int i = blockIdx.x * blockDim.x + threadIdx.x;
  if (i < n) p[i] = v;
}
__global__ void fill_i32(int* p, int v, int n) {
  int i = blockIdx.x * blockDim.x + threadIdx.x;
  if (i < n) p[i] = v;
}

__global__ void edge_pass1(const int* __restrict__ ei, int E, int N,
                           const float* __restrict__ aS, const float* __restrict__ aD,
                           float* __restrict__ eTmp, int* __restrict__ segMax)
{
  int e = blockIdx.x * blockDim.x + threadIdx.x;
  int Et = E + N;
  if (e >= Et) return;
  int s, d;
  if (e < E) { s = ei[e]; d = ei[E + e]; } else { s = d = e - E; }
  float v = aS[s] + aD[d];
  v = v > 0.f ? v : 0.2f * v;   // leaky_relu(0.2)
  eTmp[e] = v;
  atomicMax(&segMax[d], f2ord(v));
}

__global__ void edge_pass2(const int* __restrict__ ei, int E, int N,
                           float* __restrict__ eTmp,
                           const int* __restrict__ segMax, float* __restrict__ denom)
{
  int e = blockIdx.x * blockDim.x + threadIdx.x;
  int Et = E + N;
  if (e >= Et) return;
  int d = (e < E) ? ei[E + e] : (e - E);
  float ex = __expf(eTmp[e] - ord2f(segMax[d]));
  eTmp[e] = ex;
  atomicAdd(&denom[d], ex);
}

__global__ void edge_aggregate(const int* __restrict__ ei, int E, int N,
                               const float* __restrict__ eTmp, const float* __restrict__ denom,
                               const float* __restrict__ Hm, float* __restrict__ Out, int C)
{
  int lane = threadIdx.x & 31;
  int e = (int)((blockIdx.x * blockDim.x + threadIdx.x) >> 5);
  int Et = E + N;
  if (e >= Et) return;
  int s, d;
  if (e < E) { s = ei[e]; d = ei[E + e]; } else { s = d = e - E; }
  float coef = eTmp[e] / (denom[d] + 1e-16f);
  const float* hp = Hm + (size_t)s * C;
  float* op = Out + (size_t)d * C;
  for (int j = lane; j < C; j += 32) atomicAdd(&op[j], coef * hp[j]);
}

__global__ void post_relu_bf16(const float* __restrict__ acc, const float* __restrict__ bias,
                               u16* __restrict__ outH, int total, int C)
{
  int i = blockIdx.x * blockDim.x + threadIdx.x;
  if (i >= total) return;
  float v = acc[i] + bias[i % C];
  outH[i] = f2bf(fmaxf(v, 0.f));
}

__global__ void post_bias(float* __restrict__ out, const float* __restrict__ bias,
                          int total, int C)
{
  int i = blockIdx.x * blockDim.x + threadIdx.x;
  if (i >= total) return;
  out[i] += bias[i % C];
}

__global__ void f32_to_bf16_kernel(const float* __restrict__ in, u16* __restrict__ out, int n)
{
  int i = blockIdx.x * blockDim.x + threadIdx.x;
  if (i < n) out[i] = f2bf(in[i]);
}

// ---------------------------------------------------------------------------
// Host launch
// ---------------------------------------------------------------------------
extern "C" void kernel_launch(void* const* d_in, const int* in_sizes, int n_in,
                              void* d_out, int out_size, void* d_ws, size_t ws_size,
                              hipStream_t stream)
{
  (void)n_in; (void)out_size; (void)ws_size;
  const int S = 32, Dm = 128, HID = 512, OUTC = 512, F = 384;
  const int N = in_sizes[0] / (S * Dm);
  const int E = in_sizes[6] / 2;

  const float* bp_in = (const float*)d_in[0];
  const unsigned char* bp_mask = (const unsigned char*)d_in[1];
  const float* mf_in = (const float*)d_in[2];
  const unsigned char* mf_mask = (const unsigned char*)d_in[3];
  const float* cc_in = (const float*)d_in[4];
  const unsigned char* cc_mask = (const unsigned char*)d_in[5];
  const int* edge_index = (const int*)d_in[6];
  const float* bpP[13]; const float* ccP[13];
  for (int i = 0; i < 13; ++i) { bpP[i] = (const float*)d_in[7 + i]; ccP[i] = (const float*)d_in[20 + i]; }
  const float* g1W = (const float*)d_in[33];
  const float* g1as = (const float*)d_in[34];
  const float* g1ad = (const float*)d_in[35];
  const float* g1b = (const float*)d_in[36];
  const float* g2W = (const float*)d_in[37];
  const float* g2as = (const float*)d_in[38];
  const float* g2ad = (const float*)d_in[39];
  const float* g2b = (const float*)d_in[40];

  // ---- workspace carve ----
  char* ws = (char*)d_ws; size_t off = 0;
  auto carve = [&](size_t bytes) -> void* {
    off = (off + 255) & ~(size_t)255;
    void* p = ws + off; off += bytes; return p;
  };
  u16* wBp    = (u16*)carve(98304 * 2);            // Wqkv|Wo|W1|W2 bf16
  u16* wCc    = (u16*)carve(98304 * 2);
  u16* featsH = (u16*)carve((size_t)N * F * 2);
  u16* g1WH   = (u16*)carve((size_t)HID * F * 2);
  u16* g2WH   = (u16*)carve((size_t)OUTC * HID * 2);
  float* hF   = (float*)carve((size_t)N * HID * 4);
  float* accF = (float*)carve((size_t)N * HID * 4);
  u16* x2H    = (u16*)carve((size_t)N * HID * 2);
  float* alphaS = (float*)carve((size_t)N * 4);
  float* alphaD = (float*)carve((size_t)N * 4);
  int*   segMax = (int*)carve((size_t)N * 4);
  float* denom  = (float*)carve((size_t)N * 4);
  float* eTmp   = (float*)carve((size_t)(E + N) * 4);

  auto cvt = [&](const float* src, u16* dst, int n) {
    f32_to_bf16_kernel<<<(n + 255) / 256, 256, 0, stream>>>(src, dst, n);
  };
  cvt(bpP[0], wBp,          384 * 128);
  cvt(bpP[2], wBp + 49152,  128 * 128);
  cvt(bpP[6], wBp + 65536,  128 * 128);
  cvt(bpP[8], wBp + 81920,  128 * 128);
  cvt(ccP[0], wCc,          384 * 128);
  cvt(ccP[2], wCc + 49152,  128 * 128);
  cvt(ccP[6], wCc + 65536,  128 * 128);
  cvt(ccP[8], wCc + 81920,  128 * 128);
  cvt(g1W, g1WH, HID * F);
  cvt(g2W, g2WH, OUTC * HID);

  // ---- encoders (note: mf uses bp params, per reference source bug) ----
  const size_t smem = ENC_SMEM_BYTES;
  encoder_kernel<<<N, 256, smem, stream>>>(bp_in, bp_mask,
      wBp, wBp + 49152, wBp + 65536, wBp + 81920,
      bpP[1], bpP[3], bpP[7], bpP[9], bpP[4], bpP[5], bpP[10], bpP[11], bpP[12],
      featsH, F, 0);
  encoder_kernel<<<N, 256, smem, stream>>>(mf_in, mf_mask,
      wBp, wBp + 49152, wBp + 65536, wBp + 81920,
      bpP[1], bpP[3], bpP[7], bpP[9], bpP[4], bpP[5], bpP[10], bpP[11], bpP[12],
      featsH, F, 128);
  encoder_kernel<<<N, 256, smem, stream>>>(cc_in, cc_mask,
      wCc, wCc + 49152, wCc + 65536, wCc + 81920,
      ccP[1], ccP[3], ccP[7], ccP[9], ccP[4], ccP[5], ccP[10], ccP[11], ccP[12],
      featsH, F, 256);

  float ninf = -3.0e38f; int bi; memcpy(&bi, &ninf, sizeof(bi));
  const int encNegInf = bi >= 0 ? bi : (bi ^ 0x7fffffff);
  const int Et = E + N;

  // ---- GAT layer 1 ----
  {
    int tiles = (N / 16) * (HID / 16);
    gemm_bf16_kernel<<<(tiles + 7) / 8, 256, 0, stream>>>(featsH, g1WH, hF, N, HID, F);
    alpha_kernel<<<(N * 32 + 255) / 256, 256, 0, stream>>>(hF, g1as, g1ad, alphaS, alphaD, N, HID);
    fill_i32<<<(N + 255) / 256, 256, 0, stream>>>(segMax, encNegInf, N);
    fill_f32<<<(N + 255) / 256, 256, 0, stream>>>(denom, 0.f, N);
    fill_f32<<<(N * HID + 255) / 256, 256, 0, stream>>>(accF, 0.f, N * HID);
    edge_pass1<<<(Et + 255) / 256, 256, 0, stream>>>(edge_index, E, N, alphaS, alphaD, eTmp, segMax);
    edge_pass2<<<(Et + 255) / 256, 256, 0, stream>>>(edge_index, E, N, eTmp, segMax, denom);
    edge_aggregate<<<(Et * 32 + 255) / 256, 256, 0, stream>>>(edge_index, E, N, eTmp, denom, hF, accF, HID);
    post_relu_bf16<<<(N * HID + 255) / 256, 256, 0, stream>>>(accF, g1b, x2H, N * HID, HID);
  }

  // ---- GAT layer 2 (aggregate straight into d_out) ----
  {
    float* outF = (float*)d_out;
    int tiles = (N / 16) * (OUTC / 16);
    gemm_bf16_kernel<<<(tiles + 7) / 8, 256, 0, stream>>>(x2H, g2WH, hF, N, OUTC, HID);
    alpha_kernel<<<(N * 32 + 255) / 256, 256, 0, stream>>>(hF, g2as, g2ad, alphaS, alphaD, N, OUTC);
    fill_i32<<<(N + 255) / 256, 256, 0, stream>>>(segMax, encNegInf, N);
    fill_f32<<<(N + 255) / 256, 256, 0, stream>>>(denom, 0.f, N);
    fill_f32<<<(N * OUTC + 255) / 256, 256, 0, stream>>>(outF, 0.f, N * OUTC);
    edge_pass1<<<(Et + 255) / 256, 256, 0, stream>>>(edge_index, E, N, alphaS, alphaD, eTmp, segMax);
    edge_pass2<<<(Et + 255) / 256, 256, 0, stream>>>(edge_index, E, N, eTmp, segMax, denom);
    edge_aggregate<<<(Et * 32 + 255) / 256, 256, 0, stream>>>(edge_index, E, N, eTmp, denom, hF, outF, OUTC);
    post_bias<<<(N * OUTC + 255) / 256, 256, 0, stream>>>(outF, g2b, N * OUTC, OUTC);
  }
}
